// LossFunction_40346922778857
// MI455X (gfx1250) — matrix operand order repaired and checked by reading, not compiled, not run on hardware
//
#include <hip/hip_runtime.h>
#include <hip/hip_bf16.h>

// ---------------------------------------------------------------------------
// Loss over scatter matrices, restructured as:
//   G    = X^T X          (X = raw 16384x768 per half)    <- big WMMA syrk
//   Gxm  = Xm^T Xm        (Xm = 128x768 per-clip means)   <- small WMMA syrk
//   within  = G/16384 - Gxm/128
//   between = Gxm/128 - m m^T
// fp32 emulated via 3x bf16 WMMA (hi*hi + hi*lo + lo*hi), K=32 per issue.
// Double-buffered LDS panels; b128 staging loads; one barrier per K-step.
// ---------------------------------------------------------------------------

typedef __bf16 bf16t;
typedef bf16t v16bf __attribute__((ext_vector_type(16)));
typedef bf16t v8bf  __attribute__((ext_vector_type(8)));
typedef float v8f   __attribute__((ext_vector_type(8)));
typedef float v4f   __attribute__((ext_vector_type(4)));

union V16 { v16bf v; v8bf h[2]; };

static constexpr int kD      = 768;             // feature dim
static constexpr int kN      = 128;             // frames per clip
static constexpr int kBH     = 128;             // clips per half
static constexpr int kRows   = kBH * kN;        // 16384 samples per half
static constexpr int kDD     = kD * kD;         // 589824
static constexpr int kChunks = 4;               // K-split for the big gram

__device__ __forceinline__ unsigned short f2bf_rne(float f) {
  unsigned int u = __float_as_uint(f);
  u += 0x7FFFu + ((u >> 16) & 1u);              // round-to-nearest-even
  return (unsigned short)(u >> 16);
}
__device__ __forceinline__ float bfbits2f(unsigned short b) {
  return __uint_as_float(((unsigned int)b) << 16);
}

// ---------------------------------------------------------------------------
// K1: xmean[h][b][d] = mean_n x[(2b+h)][n][d]
// ---------------------------------------------------------------------------
__global__ __launch_bounds__(256) void mean_kernel(const float* __restrict__ x,
                                                   float* __restrict__ xmean) {
  int idx = blockIdx.x * 256 + threadIdx.x;     // 2*128*768 = 196608 threads
  int d = idx % kD;
  int b = (idx / kD) % kBH;
  int h = idx / (kD * kBH);
  const float* p = x + ((size_t)((2 * b + h) * kN)) * kD + d;
  float s = 0.f;
#pragma unroll 8
  for (int n = 0; n < kN; ++n) s += p[(size_t)n * kD];
  xmean[idx] = s * (1.0f / (float)kN);
}

// ---------------------------------------------------------------------------
// K1b: gmean[h][d] = mean_b xmean[h][b][d]
// ---------------------------------------------------------------------------
__global__ __launch_bounds__(256) void gmean_kernel(const float* __restrict__ xmean,
                                                    float* __restrict__ gmean) {
  int idx = blockIdx.x * 256 + threadIdx.x;     // 2*768 = 1536 threads
  if (idx >= 2 * kD) return;
  int h = idx / kD;
  int d = idx % kD;
  const float* p = xmean + (size_t)h * kBH * kD + d;
  float s = 0.f;
#pragma unroll 8
  for (int b = 0; b < kBH; ++b) s += p[(size_t)b * kD];
  gmean[idx] = s * (1.0f / (float)kBH);
}

// ---------------------------------------------------------------------------
// K2: out[(c*2+h)] = A^T A over one K-chunk, 3x bf16 WMMA per 16x16 tile.
//   64x64 output tile / 256-thread block (8 waves; each wave: 16x32 -> 2 acc).
//   INTERLEAVED=1 : rows come from x with the [::2]/[1::2] mapping.
//   INTERLEAVED=0 : rows come from xmean (contiguous per half).
// Grid.x = kchunks * 2 * 144, 144 = (768/64)^2 tiles.
// ---------------------------------------------------------------------------
template <int INTERLEAVED>
__global__ __launch_bounds__(256)
void gram_bf16x3_kernel(const float* __restrict__ src, float* __restrict__ out,
                        const int rows_per_half, const int rows_per_chunk) {
  // Double-buffered LDS panels [buf][hi/lo][col][k]; k contiguous so every
  // fragment slice is an aligned 16B ds_load_b128. 32 KB total.
  __shared__ alignas(16) unsigned short As[2][2][64][32];
  __shared__ alignas(16) unsigned short Bs[2][2][64][32];

  const int tid  = threadIdx.x;
  const int lane = tid & 31;
  const int wid  = tid >> 5;
  const int wr   = wid & 3;                     // 16-row strip within tile
  const int wc   = wid >> 2;                    // 32-col strip within tile

  int bid = blockIdx.x;
  const int c = bid / 288;                      // K-chunk
  bid %= 288;
  const int h  = bid / 144;                     // half
  const int t  = bid % 144;
  const int ti = t / 12, tj = t % 12;
  const int dbase = ti * 64, ebase = tj * 64;

  v8f acc0 = {};
  v8f acc1 = {};

  const int r0     = c * rows_per_chunk;
  const int kiters = rows_per_chunk >> 5;       // K=32 samples per iteration
  const int c4     = (tid & 15) << 2;           // this thread's 4-col group

  // Stage one 32xK panel pair into buffer `bsel` with b128 loads.
  auto stage = [&](int bsel, int rbase, bool pf) {
#pragma unroll
    for (int s = 0; s < 2; ++s) {
      const int k = (tid >> 4) + (s << 4);      // 0..31, 16 lanes per k-row
      const int r = rbase + k;
      const int mrow = INTERLEAVED ? (r + (((r >> 7) + h) << 7))
                                   : (h * rows_per_half + r);
      const float* prow = src + (size_t)mrow * kD;
      v4f va = *(const v4f*)(prow + dbase + c4);
      v4f vb = *(const v4f*)(prow + ebase + c4);
      if (pf) {                                 // warm the panel 2 K-steps out
        __builtin_prefetch(prow + (size_t)64 * kD + dbase + c4, 0, 1);
        __builtin_prefetch(prow + (size_t)64 * kD + ebase + c4, 0, 1);
      }
#pragma unroll
      for (int j = 0; j < 4; ++j) {
        unsigned short ha = f2bf_rne(va[j]);
        As[bsel][0][c4 + j][k] = ha;
        As[bsel][1][c4 + j][k] = f2bf_rne(va[j] - bfbits2f(ha));
        unsigned short hb = f2bf_rne(vb[j]);
        Bs[bsel][0][c4 + j][k] = hb;
        Bs[bsel][1][c4 + j][k] = f2bf_rne(vb[j] - bfbits2f(hb));
      }
    }
  };

  // Fragment addressing (wave32 layouts, CDNA5 ISA 7.12.2)
  const int m    = lane & 15;
  const int kbA  = (lane < 16) ? 0 : 8;   // A: VGPR0-3 K=kbA..+7, VGPR4-7 K=16+kbA..+7
  const int kbB  = (lane < 16) ? 0 : 8;   // B: VGPR0-3 K=kbB..+7, VGPR4-7 K=16+kbB..+7
  const int acol = wr * 16 + m;
  const int bc0  = wc * 32 + m;
  const int bc1  = bc0 + 16;

  stage(0, r0, r0 + 96 <= rows_per_half);
  __syncthreads();

  for (int it = 0; it < kiters; ++it) {
    const int buf = it & 1;
    if (it + 1 < kiters) {                      // overlap next stage with WMMAs
      const int nrb = r0 + ((it + 1) << 5);
      stage(buf ^ 1, nrb, nrb + 96 <= rows_per_half);
    }

    V16 aH, aL, bH0, bL0, bH1, bL1;
    aH.h[0] = *(const v8bf*)&As[buf][0][acol][kbA];
    aH.h[1] = *(const v8bf*)&As[buf][0][acol][16 + kbA];
    aL.h[0] = *(const v8bf*)&As[buf][1][acol][kbA];
    aL.h[1] = *(const v8bf*)&As[buf][1][acol][16 + kbA];
    bH0.h[0] = *(const v8bf*)&Bs[buf][0][bc0][kbB];
    bH0.h[1] = *(const v8bf*)&Bs[buf][0][bc0][kbB + 16];
    bL0.h[0] = *(const v8bf*)&Bs[buf][1][bc0][kbB];
    bL0.h[1] = *(const v8bf*)&Bs[buf][1][bc0][kbB + 16];
    bH1.h[0] = *(const v8bf*)&Bs[buf][0][bc1][kbB];
    bH1.h[1] = *(const v8bf*)&Bs[buf][0][bc1][kbB + 16];
    bL1.h[0] = *(const v8bf*)&Bs[buf][1][bc1][kbB];
    bL1.h[1] = *(const v8bf*)&Bs[buf][1][bc1][kbB + 16];

    // D = A^T A : 3-product bf16 emulation of fp32 (lo*lo term negligible)
    acc0 = __builtin_amdgcn_wmma_f32_16x16x32_bf16(false, aH.v, false, bH0.v, (short)0, acc0, false, false);
    acc0 = __builtin_amdgcn_wmma_f32_16x16x32_bf16(false, aH.v, false, bL0.v, (short)0, acc0, false, false);
    acc0 = __builtin_amdgcn_wmma_f32_16x16x32_bf16(false, aL.v, false, bH0.v, (short)0, acc0, false, false);
    acc1 = __builtin_amdgcn_wmma_f32_16x16x32_bf16(false, aH.v, false, bH1.v, (short)0, acc1, false, false);
    acc1 = __builtin_amdgcn_wmma_f32_16x16x32_bf16(false, aH.v, false, bL1.v, (short)0, acc1, false, false);
    acc1 = __builtin_amdgcn_wmma_f32_16x16x32_bf16(false, aL.v, false, bH1.v, (short)0, acc1, false, false);
    __syncthreads();                            // single barrier per K-step
  }

  // Epilogue: C/D layout — VGPR v holds M=v (lanes 0-15) / M=v+8 (lanes 16-31)
  float* outp = out + (size_t)(c * 2 + h) * kDD;
  const int n     = lane & 15;
  const int mhalf = (lane < 16) ? 0 : 8;
  const int drow0 = dbase + wr * 16;
  const int ecol0 = ebase + wc * 32;
#pragma unroll
  for (int v = 0; v < 8; ++v) {
    const int dr = drow0 + mhalf + v;
    outp[(size_t)dr * kD + ecol0 + n]      = acc0[v];
    outp[(size_t)dr * kD + ecol0 + 16 + n] = acc1[v];
  }
}

// ---------------------------------------------------------------------------
// K4: diag norms  norms[h*2+0] = ||diag(within_h)||, [h*2+1] = ||diag(between_h)||
// ---------------------------------------------------------------------------
__global__ __launch_bounds__(256)
void norms_kernel(const float* __restrict__ Gp, const float* __restrict__ Gxm,
                  const float* __restrict__ gmean, float* __restrict__ norms) {
  const int h = blockIdx.x;
  float sw = 0.f, sb = 0.f;
  for (int d = threadIdx.x; d < kD; d += 256) {
    const size_t di = (size_t)d * kD + d;
    float g = 0.f;
#pragma unroll
    for (int cc = 0; cc < kChunks; ++cc) g += Gp[(size_t)(cc * 2 + h) * kDD + di];
    const float gx = Gxm[(size_t)h * kDD + di];
    const float mm = gmean[h * kD + d];
    const float w  = g * (1.0f / (float)kRows) - gx * (1.0f / (float)kBH);
    const float bt = gx * (1.0f / (float)kBH) - mm * mm;
    sw += w * w;
    sb += bt * bt;
  }
  __shared__ float rw[256], rb[256];
  rw[threadIdx.x] = sw; rb[threadIdx.x] = sb;
  __syncthreads();
  for (int s = 128; s > 0; s >>= 1) {
    if (threadIdx.x < s) { rw[threadIdx.x] += rw[threadIdx.x + s]; rb[threadIdx.x] += rb[threadIdx.x + s]; }
    __syncthreads();
  }
  if (threadIdx.x == 0) {
    norms[h * 2 + 0] = sqrtf(rw[0]);
    norms[h * 2 + 1] = sqrtf(rb[0]);
  }
}

// ---------------------------------------------------------------------------
// K5: per-block partial sums of the normalized squared differences
// ---------------------------------------------------------------------------
__global__ __launch_bounds__(256)
void diff_partial_kernel(const float* __restrict__ Gp, const float* __restrict__ Gxm,
                         const float* __restrict__ gmean, const float* __restrict__ norms,
                         float* __restrict__ partials) {
  const int i = blockIdx.x * 256 + threadIdx.x; // 2304*256 == 589824 exactly
  const int d = i / kD, e = i % kD;
  float g0 = 0.f, g1 = 0.f;
#pragma unroll
  for (int cc = 0; cc < kChunks; ++cc) {
    g0 += Gp[(size_t)(cc * 2 + 0) * kDD + i];
    g1 += Gp[(size_t)(cc * 2 + 1) * kDD + i];
  }
  const float gx0 = Gxm[i];
  const float gx1 = Gxm[(size_t)kDD + i];
  const float w1 = g0 * (1.0f / (float)kRows) - gx0 * (1.0f / (float)kBH);
  const float w2 = g1 * (1.0f / (float)kRows) - gx1 * (1.0f / (float)kBH);
  const float b1 = gx0 * (1.0f / (float)kBH) - gmean[d] * gmean[e];
  const float b2 = gx1 * (1.0f / (float)kBH) - gmean[kD + d] * gmean[kD + e];
  const float dw = w1 / norms[0] - w2 / norms[2];
  const float db = b1 / norms[1] - b2 / norms[3];
  float val = dw * dw + db * db;                // WEIGHT_IN = WEIGHT_BET = 1

  __shared__ float r[256];
  r[threadIdx.x] = val;
  __syncthreads();
  for (int s = 128; s > 0; s >>= 1) {
    if (threadIdx.x < s) r[threadIdx.x] += r[threadIdx.x + s];
    __syncthreads();
  }
  if (threadIdx.x == 0) partials[blockIdx.x] = r[0];
}

// ---------------------------------------------------------------------------
// K6: deterministic final reduction of 2304 partials -> d_out[0]
// ---------------------------------------------------------------------------
__global__ __launch_bounds__(256)
void final_reduce_kernel(const float* __restrict__ partials, float* __restrict__ out) {
  float s = 0.f;
  for (int i = threadIdx.x; i < 2304; i += 256) s += partials[i];
  __shared__ float r[256];
  r[threadIdx.x] = s;
  __syncthreads();
  for (int st = 128; st > 0; st >>= 1) {
    if (threadIdx.x < st) r[threadIdx.x] += r[threadIdx.x + st];
    __syncthreads();
  }
  if (threadIdx.x == 0) out[0] = r[0];
}

// ---------------------------------------------------------------------------
extern "C" void kernel_launch(void* const* d_in, const int* in_sizes, int n_in,
                              void* d_out, int out_size, void* d_ws, size_t ws_size,
                              hipStream_t stream) {
  (void)in_sizes; (void)n_in; (void)out_size; (void)ws_size;
  const float* x = (const float*)d_in[0];       // (256,128,768) fp32; labels unused
  float* out = (float*)d_out;

  float* ws       = (float*)d_ws;               // ~24.4 MB of fp32 scratch
  float* xmean    = ws;                                   // 2*128*768
  float* gmean    = xmean + (size_t)2 * kBH * kD;         // 2*768
  float* Gp       = gmean + (size_t)2 * kD;               // kChunks*2*768*768
  float* Gxm      = Gp + (size_t)kChunks * 2 * kDD;       // 2*768*768
  float* norms    = Gxm + (size_t)2 * kDD;                // 4
  float* partials = norms + 4;                            // 2304

  mean_kernel<<<768, 256, 0, stream>>>(x, xmean);
  gmean_kernel<<<6, 256, 0, stream>>>(xmean, gmean);
  // big syrk: 4 K-chunks x 2 halves x 144 tiles = 1152 blocks
  gram_bf16x3_kernel<1><<<kChunks * 288, 256, 0, stream>>>(x, Gp, kRows, kRows / kChunks);
  // small syrk over the per-clip means (K = 128, one chunk)
  gram_bf16x3_kernel<0><<<288, 256, 0, stream>>>(xmean, Gxm, kBH, kBH);
  norms_kernel<<<2, 256, 0, stream>>>(Gp, Gxm, gmean, norms);
  diff_partial_kernel<<<2304, 256, 0, stream>>>(Gp, Gxm, gmean, norms, partials);
  final_reduce_kernel<<<1, 256, 0, stream>>>(partials, out);
}
